// TransformerBlock_19335942766773
// MI455X (gfx1250) — compile-verified, matrix operand order checked by reading, not run on hardware
//
#include <hip/hip_runtime.h>
#include <hip/hip_bf16.h>
#include <stdint.h>

typedef __bf16 bf16x16 __attribute__((ext_vector_type(16)));
typedef float f32x8   __attribute__((ext_vector_type(8)));

#if __has_builtin(__builtin_amdgcn_global_load_async_to_lds_b128) && \
    __has_builtin(__builtin_amdgcn_s_wait_asynccnt)
#define USE_ASYNC_LDS 1
#else
#define USE_ASYNC_LDS 0
#endif

// 128-bit payload type matching the async-LDS builtin prototype exactly
typedef int b128_t __attribute__((vector_size(16)));
typedef __attribute__((address_space(1))) b128_t* g_b128_ptr;
typedef __attribute__((address_space(3))) b128_t* l_b128_ptr;

// ---------- bf16 <-> f32 helpers (bit-exact RNE, no libcalls) ----------
static __device__ __forceinline__ __bf16 f2bf(float f) {
  union { float f; unsigned u; } a; a.f = f;
  unsigned r = a.u + 0x7FFFu + ((a.u >> 16) & 1u);
  union { unsigned short s; __bf16 b; } o; o.s = (unsigned short)(r >> 16);
  return o.b;
}
static __device__ __forceinline__ float bf2f(__bf16 b) {
  union { unsigned short s; __bf16 b; } i; i.b = b;
  union { unsigned u; float f; } o; o.u = ((unsigned)i.s) << 16;
  return o.f;
}
static __device__ __forceinline__ float gelu_f(float x) {
  float x3 = x * x * x;
  return 0.5f * x * (1.0f + tanhf(0.79788456f * (x + 0.044715f * x3)));
}

// 16-byte global -> LDS copy: async path (ASYNCcnt) if available, else via VGPRs
static __device__ __forceinline__ void cp16(__bf16* lds_dst, const __bf16* g_src) {
#if USE_ASYNC_LDS
  __builtin_amdgcn_global_load_async_to_lds_b128(
      (g_b128_ptr)(uintptr_t)g_src,
      (l_b128_ptr)(uint32_t)(uintptr_t)lds_dst, 0, 0);
#else
  *(uint4*)lds_dst = *(const uint4*)g_src;
#endif
}
static __device__ __forceinline__ void cp16_wait() {
#if USE_ASYNC_LDS
  __builtin_amdgcn_s_wait_asynccnt(0);
#endif
}

// ---------- f32 -> bf16 weight conversion (float4 -> 4x bf16) ----------
__global__ __launch_bounds__(256) void cvt_bf16_kernel(const float* __restrict__ in,
                                                       __bf16* __restrict__ out, int n) {
  int i = (blockIdx.x * 256 + threadIdx.x) * 4;
  if (i < n) {
    float4 v = *(const float4*)(in + i);
    union { uint2 u; __bf16 b[4]; } o;
    o.b[0] = f2bf(v.x); o.b[1] = f2bf(v.y); o.b[2] = f2bf(v.z); o.b[3] = f2bf(v.w);
    *(uint2*)(out + i) = o.u;
  }
}

// ---------- LayerNorm over D=1024 (4 elems/thread, single-pass moments) ----------
__global__ __launch_bounds__(256) void ln_bf16_kernel(const float* __restrict__ x,
                                                      const float* __restrict__ sc,
                                                      const float* __restrict__ sh,
                                                      __bf16* __restrict__ out, int Dn) {
  __shared__ float red[256], red2[256];
  const int tid = threadIdx.x;
  const float* xr = x + (size_t)blockIdx.x * Dn;
  float4 xv = *(const float4*)(xr + tid * 4);
  red[tid]  = xv.x + xv.y + xv.z + xv.w;
  red2[tid] = xv.x * xv.x + xv.y * xv.y + xv.z * xv.z + xv.w * xv.w;
  __syncthreads();
  for (int st = 128; st > 0; st >>= 1) {
    if (tid < st) { red[tid] += red[tid + st]; red2[tid] += red2[tid + st]; }
    __syncthreads();
  }
  float mean = red[0] / (float)Dn;
  float var  = red2[0] / (float)Dn - mean * mean;
  float rstd = rsqrtf(var + 1e-5f);
  float4 scv = *(const float4*)(sc + tid * 4);
  float4 shv = *(const float4*)(sh + tid * 4);
  union { uint2 u; __bf16 b[4]; } o;
  o.b[0] = f2bf(scv.x * (xv.x - mean) * rstd + shv.x);
  o.b[1] = f2bf(scv.y * (xv.y - mean) * rstd + shv.y);
  o.b[2] = f2bf(scv.z * (xv.z - mean) * rstd + shv.z);
  o.b[3] = f2bf(scv.w * (xv.w - mean) * rstd + shv.w);
  *(uint2*)(out + (size_t)blockIdx.x * Dn + tid * 4) = o.u;
}

// ---------- bf16 WMMA GEMM: C = A(MxK) * B(KxN) [+epilogue] ----------
// EPI 0: store bf16   EPI 1: f32 = acc + bias + residual   EPI 2: bf16 = gelu(acc+bias)
#define BM 128
#define BN 64
#define BKK 32
template <int EPI>
__global__ __launch_bounds__(256) void gemm_bf16_kernel(const __bf16* __restrict__ A,
                                                        const __bf16* __restrict__ Bm,
                                                        const float* __restrict__ bias,
                                                        const float* __restrict__ res,
                                                        float* __restrict__ outf,
                                                        __bf16* __restrict__ outb,
                                                        int M, int N, int K) {
  __shared__ alignas(16) __bf16 As[BM][BKK + 8];   // 10 KB (row stride 80B, 16B aligned)
  __shared__ alignas(16) __bf16 Bs[BKK][BN + 8];   // 4.5 KB (row stride 144B, 16B aligned)
  const int tid = threadIdx.x, lane = tid & 31, wave = tid >> 5;
  const int wm = wave * 16;                  // each wave owns a 16-row strip, all 64 cols
  const int m0 = blockIdx.y * BM, n0 = blockIdx.x * BN;
  const int lrow16 = lane & 15;
  const int lkb   = (lane >> 4) * 8;         // K split across half-waves
  const int rbase = (lane >> 4) * 8;         // accumulator row offset
  // A staging: 512 x 16B chunks, 2/thread ; B staging: 256 x 16B chunks, 1/thread
  const int ar0 = tid >> 2, ar1 = ar0 + 64, acol = (tid & 3) * 8;
  const int br = tid >> 3, bcol = (tid & 7) * 8;
  f32x8 acc[4] = {};

  for (int k0 = 0; k0 < K; k0 += BKK) {
    cp16(&As[ar0][acol], A + (size_t)(m0 + ar0) * K + k0 + acol);
    cp16(&As[ar1][acol], A + (size_t)(m0 + ar1) * K + k0 + acol);
    cp16(&Bs[br][bcol], Bm + (size_t)(k0 + br) * N + n0 + bcol);
    cp16_wait();
    __syncthreads();
    bf16x16 a;
#pragma unroll
    for (int e = 0; e < 8; ++e) {
      a[e]     = As[wm + lrow16][lkb + e];
      a[e + 8] = As[wm + lrow16][lkb + 16 + e];
    }
#pragma unroll
    for (int t = 0; t < 4; ++t) {
      bf16x16 b;
#pragma unroll
      for (int e = 0; e < 8; ++e) {
        b[e]     = Bs[lkb + e][t * 16 + lrow16];
        b[e + 8] = Bs[lkb + 16 + e][t * 16 + lrow16];
      }
      acc[t] = __builtin_amdgcn_wmma_f32_16x16x32_bf16(false, a, false, b, (short)0, acc[t], false, false);
    }
    __syncthreads();
  }

#pragma unroll
  for (int t = 0; t < 4; ++t) {
    const int c = n0 + t * 16 + lrow16;
#pragma unroll
    for (int r = 0; r < 8; ++r) {
      size_t i = (size_t)(m0 + wm + rbase + r) * N + c;
      float v = acc[t][r];
      if constexpr (EPI == 0) {
        outb[i] = f2bf(v);
      } else if constexpr (EPI == 1) {
        outf[i] = v + bias[c] + res[i];
      } else {
        outb[i] = f2bf(gelu_f(v + bias[c]));
      }
    }
  }
}

// ---------- causal flash attention, WMMA for QK^T and PV ----------
#define QT 64
#define KT 32
__global__ __launch_bounds__(256) void flash_attn_kernel(const __bf16* __restrict__ q,
                                                         const __bf16* __restrict__ k,
                                                         const __bf16* __restrict__ v,
                                                         __bf16* __restrict__ ctx,
                                                         int S, int Dm, int Hn, int HDn) {
  __shared__ alignas(16) __bf16 Qs[QT][64 + 8];
  __shared__ alignas(16) __bf16 Ks[KT][64 + 8];
  __shared__ alignas(16) __bf16 Vs[KT][64 + 8];
  __shared__ float Sb[QT][KT + 1];
  __shared__ alignas(16) __bf16 Pb[QT][KT + 8];
  __shared__ float mrow[QT], lrow[QT], arow[QT];

  const int tid = threadIdx.x, lane = tid & 31, wave = tid >> 5;
  const int bh = blockIdx.y, bb = bh / Hn, hh = bh % Hn;
  const int q0 = blockIdx.x * QT;
  const size_t base = (size_t)bb * S * Dm + (size_t)hh * HDn;

  // Q tile, pre-scaled by 1/sqrt(HD) = 0.125 (8 bf16 per 16B chunk)
  for (int c = tid; c < QT * 8; c += 256) {
    int r = c >> 3, cc = (c & 7) * 8;
    union { uint4 u; __bf16 b[8]; } t;
    t.u = *(const uint4*)(q + base + (size_t)(q0 + r) * Dm + cc);
#pragma unroll
    for (int e = 0; e < 8; ++e) t.b[e] = f2bf(bf2f(t.b[e]) * 0.125f);
    *(uint4*)&Qs[r][cc] = t.u;
  }
  if (tid < QT) { mrow[tid] = -1e30f; lrow[tid] = 0.f; }

  const int lrow16 = lane & 15;
  const int lkb = (lane >> 4) * 8;
  const int rbase = (lane >> 4) * 8;
  const int sqb = (wave & 3) * 16, skb = (wave >> 2) * 16;  // S tile (64x32)
  const int oqb = (wave & 3) * 16, onb = (wave >> 2) * 32;  // O tile (64x64)
  const int kvr = tid >> 3, kvc = (tid & 7) * 8;            // K/V staging chunk
  f32x8 o0 = {}, o1 = {};
  __syncthreads();

  for (int k0 = 0; k0 <= q0 + QT - 1; k0 += KT) {
    {
      size_t g = base + (size_t)(k0 + kvr) * Dm + kvc;
      cp16(&Ks[kvr][kvc], k + g);
      cp16(&Vs[kvr][kvc], v + g);
      cp16_wait();
    }
    __syncthreads();

    // S = Q * K^T (two 32-wide HD chunks)
    f32x8 s = {};
#pragma unroll
    for (int kc = 0; kc < 2; ++kc) {
      int kb = kc * 32;
      bf16x16 a, b;
#pragma unroll
      for (int e = 0; e < 8; ++e) {
        a[e]     = Qs[sqb + lrow16][kb + lkb + e];
        a[e + 8] = Qs[sqb + lrow16][kb + lkb + 16 + e];
        b[e]     = Ks[skb + lrow16][kb + lkb + e];
        b[e + 8] = Ks[skb + lrow16][kb + lkb + 16 + e];
      }
      s = __builtin_amdgcn_wmma_f32_16x16x32_bf16(false, a, false, b, (short)0, s, false, false);
    }
#pragma unroll
    for (int r = 0; r < 8; ++r) Sb[sqb + rbase + r][skb + lrow16] = s[r];
    __syncthreads();

    // online softmax: 4 threads per row, 8 cols each, shfl within wave32
    {
      const int row = tid >> 2, cc0 = (tid & 3) * 8, qg = q0 + row;
      float mx = -1e30f, sv[8];
#pragma unroll
      for (int e = 0; e < 8; ++e) {
        float val = (k0 + cc0 + e <= qg) ? Sb[row][cc0 + e] : -1e30f;
        sv[e] = val; mx = fmaxf(mx, val);
      }
      mx = fmaxf(mx, __shfl_xor(mx, 1, 32));
      mx = fmaxf(mx, __shfl_xor(mx, 2, 32));
      float mold = mrow[row];
      float mnew = fmaxf(mold, mx);
      float sum = 0.f;
#pragma unroll
      for (int e = 0; e < 8; ++e) {
        float p = (k0 + cc0 + e <= qg) ? __expf(sv[e] - mnew) : 0.f;
        Pb[row][cc0 + e] = f2bf(p);
        sum += p;
      }
      sum += __shfl_xor(sum, 1, 32);
      sum += __shfl_xor(sum, 2, 32);
      if ((tid & 3) == 0) {
        float al = __expf(mold - mnew);
        arow[row] = al; mrow[row] = mnew;
        lrow[row] = lrow[row] * al + sum;
      }
    }
    __syncthreads();

    // O = O*alpha + P*V (per-row rescale applied to accumulator registers)
    {
#pragma unroll
      for (int r = 0; r < 8; ++r) {
        float al = arow[oqb + rbase + r];
        o0[r] *= al; o1[r] *= al;
      }
      bf16x16 a, b0, b1;
#pragma unroll
      for (int e = 0; e < 8; ++e) {
        a[e]     = Pb[oqb + lrow16][lkb + e];
        a[e + 8] = Pb[oqb + lrow16][lkb + 16 + e];
        b0[e]     = Vs[lkb + e][onb + lrow16];
        b0[e + 8] = Vs[lkb + 16 + e][onb + lrow16];
        b1[e]     = Vs[lkb + e][onb + 16 + lrow16];
        b1[e + 8] = Vs[lkb + 16 + e][onb + 16 + lrow16];
      }
      o0 = __builtin_amdgcn_wmma_f32_16x16x32_bf16(false, a, false, b0, (short)0, o0, false, false);
      o1 = __builtin_amdgcn_wmma_f32_16x16x32_bf16(false, a, false, b1, (short)0, o1, false, false);
    }
    __syncthreads();
  }

#pragma unroll
  for (int r = 0; r < 8; ++r) {
    int row = oqb + rbase + r;
    float inv = 1.f / lrow[row];
    size_t g = base + (size_t)(q0 + row) * Dm + onb + lrow16;
    ctx[g] = f2bf(o0[r] * inv);
    ctx[g + 16] = f2bf(o1[r] * inv);
  }
}

// ---------------------------------------------------------------
extern "C" void kernel_launch(void* const* d_in, const int* in_sizes, int n_in,
                              void* d_out, int out_size, void* d_ws, size_t ws_size,
                              hipStream_t stream) {
  (void)in_sizes; (void)n_in; (void)out_size; (void)ws_size;
  const int Bb = 2, S = 2048, D = 1024, H = 16, HD = 64, D4 = 4096;
  const int BS = Bb * S;  // 4096 rows

  const float* x   = (const float*)d_in[0];
  const float* sc1 = (const float*)d_in[1];
  const float* sh1 = (const float*)d_in[2];
  const float* Wq  = (const float*)d_in[3];
  const float* Wk  = (const float*)d_in[4];
  const float* Wv  = (const float*)d_in[5];
  const float* Wo  = (const float*)d_in[6];
  const float* bo  = (const float*)d_in[7];
  const float* sc2 = (const float*)d_in[8];
  const float* sh2 = (const float*)d_in[9];
  const float* W1  = (const float*)d_in[10];
  const float* b1  = (const float*)d_in[11];
  const float* W2  = (const float*)d_in[12];
  const float* b2  = (const float*)d_in[13];
  float* out = (float*)d_out;

  char* ws = (char*)d_ws;
  size_t off = 0;
  auto alloc = [&](size_t bytes) -> void* {
    void* p = ws + off;
    off += (bytes + 255) & ~(size_t)255;
    return p;
  };
  __bf16* Wq_b = (__bf16*)alloc((size_t)D * D * 2);
  __bf16* Wk_b = (__bf16*)alloc((size_t)D * D * 2);
  __bf16* Wv_b = (__bf16*)alloc((size_t)D * D * 2);
  __bf16* Wo_b = (__bf16*)alloc((size_t)D * D * 2);
  __bf16* W1_b = (__bf16*)alloc((size_t)D * D4 * 2);
  __bf16* W2_b = (__bf16*)alloc((size_t)D4 * D * 2);
  __bf16* h_b  = (__bf16*)alloc((size_t)BS * D * 2);
  __bf16* q_b  = (__bf16*)alloc((size_t)BS * D * 2);
  __bf16* k_b  = (__bf16*)alloc((size_t)BS * D * 2);
  __bf16* v_b  = (__bf16*)alloc((size_t)BS * D * 2);
  __bf16* ctx_b= (__bf16*)alloc((size_t)BS * D * 2);
  float*  x2   = (float*) alloc((size_t)BS * D * 4);
  __bf16* h2_b = (__bf16*)alloc((size_t)BS * D * 2);
  __bf16* h3_b = (__bf16*)alloc((size_t)BS * D4 * 2);

  // weight conversion (4 elems/thread)
  cvt_bf16_kernel<<<(D * D / 4 + 255) / 256, 256, 0, stream>>>(Wq, Wq_b, D * D);
  cvt_bf16_kernel<<<(D * D / 4 + 255) / 256, 256, 0, stream>>>(Wk, Wk_b, D * D);
  cvt_bf16_kernel<<<(D * D / 4 + 255) / 256, 256, 0, stream>>>(Wv, Wv_b, D * D);
  cvt_bf16_kernel<<<(D * D / 4 + 255) / 256, 256, 0, stream>>>(Wo, Wo_b, D * D);
  cvt_bf16_kernel<<<(D * D4 / 4 + 255) / 256, 256, 0, stream>>>(W1, W1_b, D * D4);
  cvt_bf16_kernel<<<(D4 * D / 4 + 255) / 256, 256, 0, stream>>>(W2, W2_b, D4 * D);

  // LN1 -> h
  ln_bf16_kernel<<<BS, 256, 0, stream>>>(x, sc1, sh1, h_b, D);

  // QKV projections (WMMA)
  dim3 gqkv(D / BN, BS / BM);
  gemm_bf16_kernel<0><<<gqkv, 256, 0, stream>>>(h_b, Wq_b, nullptr, nullptr, nullptr, q_b, BS, D, D);
  gemm_bf16_kernel<0><<<gqkv, 256, 0, stream>>>(h_b, Wk_b, nullptr, nullptr, nullptr, k_b, BS, D, D);
  gemm_bf16_kernel<0><<<gqkv, 256, 0, stream>>>(h_b, Wv_b, nullptr, nullptr, nullptr, v_b, BS, D, D);

  // causal flash attention
  flash_attn_kernel<<<dim3(S / QT, Bb * H), 256, 0, stream>>>(q_b, k_b, v_b, ctx_b, S, D, H, HD);

  // x2 = ctx@Wo + bo + x
  gemm_bf16_kernel<1><<<dim3(D / BN, BS / BM), 256, 0, stream>>>(ctx_b, Wo_b, bo, x, x2, nullptr, BS, D, D);

  // LN2 -> h2
  ln_bf16_kernel<<<BS, 256, 0, stream>>>(x2, sc2, sh2, h2_b, D);

  // h3 = gelu(h2@W1 + b1)
  gemm_bf16_kernel<2><<<dim3(D4 / BN, BS / BM), 256, 0, stream>>>(h2_b, W1_b, b1, nullptr, nullptr, h3_b, BS, D4, D);

  // out = h3@W2 + b2 + x2
  gemm_bf16_kernel<1><<<dim3(D / BN, BS / BM), 256, 0, stream>>>(h3_b, W2_b, b2, x2, out, nullptr, BS, D, D4);
}
